// ThreeWayAttention_5394478924256
// MI455X (gfx1250) — compile-verified, matrix operand order checked by reading, not run on hardware
//
#include <hip/hip_runtime.h>
#include <hip/hip_bf16.h>

// ---------------------------------------------------------------------------
// Three-way attention for MI455X (gfx1250, wave32, WMMA).
// All matmuls use v_wmma_f32_16x16x32_f16. Every WMMA operand is stored
// K-contiguous in LDS so each fragment is exactly two ds_load_b128 per lane.
// Staging is vectorized (b128 global loads, packed LDS stores). The boolean
// mask is pre-packed into 32 VGPRs of bitmasks per lane (16 slices x 64
// positions) with a 1-mul/dword gather, removing all VMEM traffic from the
// slice loop.
// ---------------------------------------------------------------------------

typedef __attribute__((ext_vector_type(16))) _Float16 v16h;
typedef __attribute__((ext_vector_type(8)))  _Float16 h8;
typedef __attribute__((ext_vector_type(4)))  _Float16 h4;
typedef __attribute__((ext_vector_type(8)))  float    v8f;
typedef __attribute__((ext_vector_type(4)))  float    f4;
typedef __attribute__((ext_vector_type(4)))  unsigned int u32x4;

#define BSZ    2
#define NSEQ   128
#define CINCH  256
#define HEADS  8
#define DHEAD  64
#define INNERD 512
#define SIMSCALE (1.0f / 192.0f)   // (1/64)/3
#define KSPLIT 8                   // slice groups per (e,h) for occupancy
#define KPER   (NSEQ / KSPLIT)     // 16 slices per block

// Pass-2 dynamic LDS:
//  halfs : a,b,c,vc (4*8192) + vaT,vbT (2*8192) + E (16384) + ET (16384)
//        = 81920 halfs = 163840 B
//  floats: FT 8192 + wa 128 + wb 128 + outc_part 256 = 8704 = 34816 B
#define PASS2_LDS_BYTES 198656

// ---- WMMA helpers ----------------------------------------------------------

__device__ __forceinline__ v8f wmma_f16(v16h a, v16h b, v8f c) {
  return __builtin_amdgcn_wmma_f32_16x16x32_f16(false, a, false, b,
                                                (short)0, c, false, false);
}

// Fragment of a K-contiguous (row-major-in-K) operand: logical rows row0..+15,
// K window [k0, k0+32). Two 16-byte LDS loads per lane. Requires ld % 8 == 0.
__device__ __forceinline__ v16h frag_rowk(const _Float16* s, int ld,
                                          int row0, int k0, int lane) {
  const int m  = row0 + (lane & 15);
  const int hi8 = (lane >> 4) * 8;
  const _Float16* base = s + m * ld + k0 + hi8;
  v16h f;
  *((h8*)&f)     = *(const h8*)(base);
  *((h8*)&f + 1) = *(const h8*)(base + 16);
  return f;
}

// Gather 4 bool bytes (0/1) of a dword into bits 3:0 with one multiply.
// Source bits {0,8,16,24} x multiplier bits {7,14,21,28} land at disjoint
// positions, with 0+28=28, 8+21=29, 16+14=30, 24+7=31 and no carries.
__device__ __forceinline__ unsigned pack4(unsigned w) {
  return (w * 0x10204080u) >> 28;
}

// Pre-pack, per lane, the mask bits for this block's KPER slices over all of
// the lane's 64 (i,j) C-fragment positions: mreg[tj*4 + r/2], 16 bits per
// (tj,r) at halfword (r&1). One uint4 global load per position, once.
__device__ __forceinline__ void load_mask_bits(
    const unsigned char* mbase, int i0, int hi, int nn, int k0,
    unsigned* mreg) {
#pragma unroll
  for (int tj = 0; tj < 8; ++tj) {
    const int j = tj * 16 + nn;
#pragma unroll
    for (int r = 0; r < 8; ++r) {
      const int i = i0 + r + 8 * hi;
      const u32x4 mw =
          *(const u32x4*)(mbase + ((size_t)i * NSEQ + j) * NSEQ + k0);
      const unsigned bits = pack4(mw[0]) | (pack4(mw[1]) << 4) |
                            (pack4(mw[2]) << 8) | (pack4(mw[3]) << 12);
      const int q = tj * 4 + (r >> 1);
      if (r & 1) mreg[q] |= bits << 16;
      else       mreg[q]  = bits;
    }
  }
}

// ---- Generic WMMA GEMM: f32 in (staged to f16 LDS), 64x64 tile, 4 waves ----
// B tile is staged transposed (Bst[n][k]) so B-fragments are K-contiguous.

__global__ __launch_bounds__(128) void gemm_f16out_kernel(
    const float* __restrict__ X, const float* __restrict__ W,
    _Float16* __restrict__ out, int M, int K, int N) {
  __shared__ __align__(16) _Float16 As[64 * 40];
  __shared__ __align__(16) _Float16 Bst[64 * 40];
  const int tid = threadIdx.x, lane = tid & 31, wave = tid >> 5;
  const int row0 = blockIdx.y * 64, col0 = blockIdx.x * 64;
  const v8f vzero = {0.f, 0.f, 0.f, 0.f, 0.f, 0.f, 0.f, 0.f};
  v8f acc[4];
#pragma unroll
  for (int t = 0; t < 4; ++t) acc[t] = vzero;

  for (int kb = 0; kb < K; kb += 32) {
    __syncthreads();
    for (int idx = tid; idx < 64 * 8; idx += 128) {        // X: b128 loads
      const int m = idx >> 3, k4 = (idx & 7) * 4;
      const f4 x = *(const f4*)(X + (size_t)(row0 + m) * K + kb + k4);
      h4 y = {(_Float16)x[0], (_Float16)x[1], (_Float16)x[2], (_Float16)x[3]};
      *(h4*)(As + m * 40 + k4) = y;
    }
    for (int idx = tid; idx < 32 * 16; idx += 128) {       // W: b128 loads
      const int kk = idx >> 4, n4 = (idx & 15) * 4;
      const f4 wv4 = *(const f4*)(W + (size_t)(kb + kk) * N + col0 + n4);
#pragma unroll
      for (int u = 0; u < 4; ++u)
        Bst[(n4 + u) * 40 + kk] = (_Float16)wv4[u];        // transpose scatter
    }
    __syncthreads();
    const v16h af = frag_rowk(As, 40, wave * 16, 0, lane);
    v16h bfs[4];
#pragma unroll
    for (int t = 0; t < 4; ++t) bfs[t] = frag_rowk(Bst, 40, t * 16, 0, lane);
#pragma unroll
    for (int t = 0; t < 4; ++t) acc[t] = wmma_f16(af, bfs[t], acc[t]);
  }
  const int hi = lane >> 4, nn = lane & 15;
#pragma unroll
  for (int t = 0; t < 4; ++t)
#pragma unroll
    for (int r = 0; r < 8; ++r) {
      const int i = row0 + wave * 16 + r + 8 * hi;
      const int j = col0 + t * 16 + nn;
      out[(size_t)i * N + j] = (_Float16)acc[t][r];
    }
}

__global__ __launch_bounds__(128) void gemm_bias_f32out_kernel(
    const float* __restrict__ X, const float* __restrict__ W,
    const float* __restrict__ bias, float* __restrict__ out,
    int M, int K, int N) {
  __shared__ __align__(16) _Float16 As[64 * 40];
  __shared__ __align__(16) _Float16 Bst[64 * 40];
  const int tid = threadIdx.x, lane = tid & 31, wave = tid >> 5;
  const int row0 = blockIdx.y * 64, col0 = blockIdx.x * 64;
  const v8f vzero = {0.f, 0.f, 0.f, 0.f, 0.f, 0.f, 0.f, 0.f};
  v8f acc[4];
#pragma unroll
  for (int t = 0; t < 4; ++t) acc[t] = vzero;

  for (int kb = 0; kb < K; kb += 32) {
    __syncthreads();
    for (int idx = tid; idx < 64 * 8; idx += 128) {
      const int m = idx >> 3, k4 = (idx & 7) * 4;
      const f4 x = *(const f4*)(X + (size_t)(row0 + m) * K + kb + k4);
      h4 y = {(_Float16)x[0], (_Float16)x[1], (_Float16)x[2], (_Float16)x[3]};
      *(h4*)(As + m * 40 + k4) = y;
    }
    for (int idx = tid; idx < 32 * 16; idx += 128) {
      const int kk = idx >> 4, n4 = (idx & 15) * 4;
      const f4 wv4 = *(const f4*)(W + (size_t)(kb + kk) * N + col0 + n4);
#pragma unroll
      for (int u = 0; u < 4; ++u)
        Bst[(n4 + u) * 40 + kk] = (_Float16)wv4[u];
    }
    __syncthreads();
    const v16h af = frag_rowk(As, 40, wave * 16, 0, lane);
    v16h bfs[4];
#pragma unroll
    for (int t = 0; t < 4; ++t) bfs[t] = frag_rowk(Bst, 40, t * 16, 0, lane);
#pragma unroll
    for (int t = 0; t < 4; ++t) acc[t] = wmma_f16(af, bfs[t], acc[t]);
  }
  const int hi = lane >> 4, nn = lane & 15;
#pragma unroll
  for (int t = 0; t < 4; ++t)
#pragma unroll
    for (int r = 0; r < 8; ++r) {
      const int i = row0 + wave * 16 + r + 8 * hi;
      const int j = col0 + t * 16 + nn;
      out[(size_t)i * N + j] = acc[t][r] + bias[j];
    }
}

// ---- Pass 1: softmax denominators along the three axes ---------------------

__global__ __launch_bounds__(256) void pass1_stats_kernel(
    const _Float16* __restrict__ fa, const _Float16* __restrict__ fb,
    const _Float16* __restrict__ fc, const unsigned char* __restrict__ mask,
    float* __restrict__ zr_part, float* __restrict__ zc_part,
    float* __restrict__ w_c) {
  __shared__ __align__(16) _Float16 a_s[NSEQ * DHEAD];
  __shared__ __align__(16) _Float16 b_s[NSEQ * DHEAD];
  __shared__ __align__(16) _Float16 c_s[NSEQ * DHEAD];
  __shared__ float zw[8];
  __shared__ float zcpart[8 * NSEQ];

  const int blk = blockIdx.x;          // eh*KSPLIT + sk
  const int eh = blk / KSPLIT, sk = blk % KSPLIT;
  const int e = eh >> 3, h = eh & 7;
  const int tid = threadIdx.x, lane = tid & 31, wave = tid >> 5;
  const int hi = lane >> 4, nn = lane & 15;
  const int hi8 = hi * 8;
  const int i0 = wave * 16;

  for (int idx = tid; idx < (NSEQ * DHEAD) / 8; idx += 256) {
    const int n = idx >> 3, d8 = (idx & 7) * 8;
    const size_t g = (size_t)(e * NSEQ + n) * INNERD + h * DHEAD + d8;
    *(h8*)(a_s + n * DHEAD + d8) = *(const h8*)(fa + g);
    *(h8*)(b_s + n * DHEAD + d8) = *(const h8*)(fb + g);
    *(h8*)(c_s + n * DHEAD + d8) = *(const h8*)(fc + g);
  }
  __syncthreads();

  // Pre-pack mask bits for this block's slices (no VMEM in the k loop).
  const unsigned char* mbase = mask + (size_t)e * NSEQ * NSEQ * NSEQ;
  unsigned mreg[32];
  load_mask_bits(mbase, i0, hi, nn, sk * KPER, mreg);

  float rowpart[8], colpart[8];
#pragma unroll
  for (int r = 0; r < 8; ++r) { rowpart[r] = 0.f; colpart[r] = 0.f; }

  // a-fragments are k-slice invariant: hoist out of everything.
  const v16h af0 = frag_rowk(a_s, DHEAD, i0, 0, lane);
  const v16h af1 = frag_rowk(a_s, DHEAD, i0, 32, lane);

  for (int kk = 0; kk < KPER; ++kk) {
    const int k = sk * KPER + kk;
    const _Float16* crow = c_s + k * DHEAD;
    const h8 c0 = *(const h8*)(crow + hi8);
    const h8 c1 = *(const h8*)(crow + 16 + hi8);
    const h8 c2 = *(const h8*)(crow + 32 + hi8);
    const h8 c3 = *(const h8*)(crow + 48 + hi8);
    float tot = 0.f;
#pragma unroll
    for (int tj = 0; tj < 8; ++tj) {
      const int j = tj * 16 + nn;
      const _Float16* brow = b_s + j * DHEAD;
      v16h bf0, bf1;  // B[d][j] = b[j][d]*c[k][d], packed f16 multiplies
      *((h8*)&bf0)     = (*(const h8*)(brow + hi8))      * c0;
      *((h8*)&bf0 + 1) = (*(const h8*)(brow + 16 + hi8)) * c1;
      *((h8*)&bf1)     = (*(const h8*)(brow + 32 + hi8)) * c2;
      *((h8*)&bf1 + 1) = (*(const h8*)(brow + 48 + hi8)) * c3;
      v8f acc = {0.f, 0.f, 0.f, 0.f, 0.f, 0.f, 0.f, 0.f};
      acc = wmma_f16(af0, bf0, acc);
      acc = wmma_f16(af1, bf1, acc);
      float cp = 0.f;
#pragma unroll
      for (int r = 0; r < 8; ++r) {
        const float s = acc[r] * SIMSCALE;
        const unsigned bit =
            (mreg[tj * 4 + (r >> 1)] >> ((r & 1) * 16 + kk)) & 1u;
        const float E = bit ? __expf(s) : 0.f;
        rowpart[r] += E;
        cp += E;
      }
      colpart[tj] += cp;
      tot += cp;
    }
    // per-slice total -> w_c[k] (one owner block per slice)
#pragma unroll
    for (int off = 16; off > 0; off >>= 1) tot += __shfl_xor(tot, off, 32);
    if (lane == 0) zw[wave] = tot;
    __syncthreads();
    if (tid == 0) {
      float z = 0.f;
#pragma unroll
      for (int wv = 0; wv < 8; ++wv) z += zw[wv];
      w_c[eh * NSEQ + k] = 1.0f / z;
    }
    __syncthreads();
  }

  // Row sums: reduce across the 16 lanes of each half-wave.
#pragma unroll
  for (int r = 0; r < 8; ++r) {
    float v = rowpart[r];
#pragma unroll
    for (int off = 8; off > 0; off >>= 1) v += __shfl_xor(v, off, 32);
    if (nn == 0) zr_part[(size_t)blk * NSEQ + i0 + r + 8 * hi] = v;
  }
  // Col sums: reduce hi halves, then across waves via LDS.
#pragma unroll
  for (int tj = 0; tj < 8; ++tj) {
    const float v = colpart[tj] + __shfl_xor(colpart[tj], 16, 32);
    if (hi == 0) zcpart[wave * NSEQ + tj * 16 + nn] = v;
  }
  __syncthreads();
  if (tid < NSEQ) {
    float zc = 0.f;
#pragma unroll
    for (int wv = 0; wv < 8; ++wv) zc += zcpart[wv * NSEQ + tid];
    zc_part[(size_t)blk * NSEQ + tid] = zc;
  }
}

__global__ __launch_bounds__(128) void finalize_stats_kernel(
    const float* __restrict__ zr_part, const float* __restrict__ zc_part,
    float* __restrict__ w_a, float* __restrict__ w_b) {
  const int eh = blockIdx.x, i = threadIdx.x;
  float sr = 0.f, sc = 0.f;
#pragma unroll
  for (int s = 0; s < KSPLIT; ++s) {
    sr += zr_part[(size_t)(eh * KSPLIT + s) * NSEQ + i];
    sc += zc_part[(size_t)(eh * KSPLIT + s) * NSEQ + i];
  }
  w_a[eh * NSEQ + i] = 1.0f / sr;
  w_b[eh * NSEQ + i] = 1.0f / sc;
}

// ---- Pass 2: attention outputs --------------------------------------------
// Per slice k: S -> E (both orientations in LDS), F = E @ vb, T = E^T @ va,
//   out_a[i,d] += w_a[i]*F*vc[k,d]   (register accumulators)
//   out_b[j,d] += w_b[j]*T*vc[k,d]   (register accumulators)
//   out_c[k,d]  = w_c[k]*sum_i va[i,d]*F[i,d]

__global__ __launch_bounds__(256) void pass2_attn_kernel(
    const _Float16* __restrict__ fa, const _Float16* __restrict__ fb,
    const _Float16* __restrict__ fc, const _Float16* __restrict__ pva,
    const _Float16* __restrict__ pvb, const _Float16* __restrict__ pvc,
    const unsigned char* __restrict__ mask,
    const float* __restrict__ w_a, const float* __restrict__ w_b,
    const float* __restrict__ w_c,
    float* __restrict__ outa_part, float* __restrict__ outb_part,
    float* __restrict__ outc) {
  extern __shared__ char smem_raw[];
  _Float16* a_s   = (_Float16*)smem_raw;        // [i][d]   128x64
  _Float16* b_s   = a_s   + 8192;               // [j][d]
  _Float16* c_s   = b_s   + 8192;               // [k][d]
  _Float16* vc_s  = c_s   + 8192;               // [k][d]
  _Float16* vaT_s = vc_s  + 8192;               // [d][i]   64x128
  _Float16* vbT_s = vaT_s + 8192;               // [d][j]
  _Float16* E_s   = vbT_s + 8192;               // [i][j]   128x128
  _Float16* ET_s  = E_s   + 16384;              // [j][i]
  float* FT_s      = (float*)(ET_s + 16384);    // [d][i]   64x128 f32
  float* wa_s      = FT_s + 8192;
  float* wb_s      = wa_s + 128;
  float* outc_part = wb_s + 128;                // 4 x 64

  const int blk = blockIdx.x;
  const int eh = blk / KSPLIT, sk = blk % KSPLIT;
  const int e = eh >> 3, h = eh & 7;
  const int tid = threadIdx.x, lane = tid & 31, wave = tid >> 5;
  const int hi = lane >> 4, nn = lane & 15;
  const int hi8 = hi * 8;
  const int i0 = wave * 16;

  for (int idx = tid; idx < (NSEQ * DHEAD) / 8; idx += 256) {
    const int n = idx >> 3, d8 = (idx & 7) * 8;
    const size_t g = (size_t)(e * NSEQ + n) * INNERD + h * DHEAD + d8;
    *(h8*)(a_s + n * DHEAD + d8)  = *(const h8*)(fa + g);
    *(h8*)(b_s + n * DHEAD + d8)  = *(const h8*)(fb + g);
    *(h8*)(c_s + n * DHEAD + d8)  = *(const h8*)(fc + g);
    *(h8*)(vc_s + n * DHEAD + d8) = *(const h8*)(pvc + g);
    const h8 va8 = *(const h8*)(pva + g);
    const h8 vb8 = *(const h8*)(pvb + g);
#pragma unroll
    for (int u = 0; u < 8; ++u) {
      vaT_s[(d8 + u) * NSEQ + n] = va8[u];      // transpose scatter
      vbT_s[(d8 + u) * NSEQ + n] = vb8[u];
    }
  }
  if (tid < NSEQ) {
    wa_s[tid] = w_a[eh * NSEQ + tid];
    wb_s[tid] = w_b[eh * NSEQ + tid];
  }
  __syncthreads();

  const unsigned char* mbase = mask + (size_t)e * NSEQ * NSEQ * NSEQ;
  unsigned mreg[32];
  load_mask_bits(mbase, i0, hi, nn, sk * KPER, mreg);

  const float* wcp = w_c + eh * NSEQ;
  const v8f vzero = {0.f, 0.f, 0.f, 0.f, 0.f, 0.f, 0.f, 0.f};
  v8f acc_a[4], acc_b[4];
#pragma unroll
  for (int t = 0; t < 4; ++t) { acc_a[t] = vzero; acc_b[t] = vzero; }

  // k-invariant a-fragments for GEMM1
  const v16h af0 = frag_rowk(a_s, DHEAD, i0, 0, lane);
  const v16h af1 = frag_rowk(a_s, DHEAD, i0, 32, lane);

  for (int kk = 0; kk < KPER; ++kk) {
    const int k = sk * KPER + kk;

    // --- GEMM1: S = a @ (b.*c_k)^T, exp -> E_s & ET_s ---
    {
      const _Float16* crow = c_s + k * DHEAD;
      const h8 c0 = *(const h8*)(crow + hi8);
      const h8 c1 = *(const h8*)(crow + 16 + hi8);
      const h8 c2 = *(const h8*)(crow + 32 + hi8);
      const h8 c3 = *(const h8*)(crow + 48 + hi8);
#pragma unroll
      for (int tj = 0; tj < 8; ++tj) {
        const int j = tj * 16 + nn;
        const _Float16* brow = b_s + j * DHEAD;
        v16h bf0, bf1;
        *((h8*)&bf0)     = (*(const h8*)(brow + hi8))      * c0;
        *((h8*)&bf0 + 1) = (*(const h8*)(brow + 16 + hi8)) * c1;
        *((h8*)&bf1)     = (*(const h8*)(brow + 32 + hi8)) * c2;
        *((h8*)&bf1 + 1) = (*(const h8*)(brow + 48 + hi8)) * c3;
        v8f acc = vzero;
        acc = wmma_f16(af0, bf0, acc);
        acc = wmma_f16(af1, bf1, acc);
        h8 ev;
#pragma unroll
        for (int r = 0; r < 8; ++r) {
          const int i = i0 + r + 8 * hi;
          const float s = acc[r] * SIMSCALE;
          const unsigned bit =
              (mreg[tj * 4 + (r >> 1)] >> ((r & 1) * 16 + kk)) & 1u;
          const float E = bit ? __expf(s) : 0.f;
          E_s[i * NSEQ + j] = (_Float16)E;   // scatter (m-strided)
          ev[r] = (_Float16)E;
        }
        *(h8*)(ET_s + j * NSEQ + i0 + 8 * hi) = ev;  // one b128 store
      }
    }
    __syncthreads();

    // --- GEMM2: F = E @ vb ; fold out_a, stash F^T for out_c ---
    {
      v8f fac[4] = {vzero, vzero, vzero, vzero};
#pragma unroll
      for (int kb = 0; kb < NSEQ; kb += 32) {
        const v16h af = frag_rowk(E_s, NSEQ, i0, kb, lane);
        v16h bfs[4];
#pragma unroll
        for (int t = 0; t < 4; ++t)
          bfs[t] = frag_rowk(vbT_s, NSEQ, t * 16, kb, lane);
#pragma unroll
        for (int t = 0; t < 4; ++t) fac[t] = wmma_f16(af, bfs[t], fac[t]);
      }
      float wv[8];
#pragma unroll
      for (int r = 0; r < 8; ++r) wv[r] = wa_s[i0 + r + 8 * hi];
#pragma unroll
      for (int t = 0; t < 4; ++t) {
        const int d = t * 16 + nn;
        const float vcv = (float)vc_s[k * DHEAD + d];
#pragma unroll
        for (int r = 0; r < 8; ++r)
          acc_a[t][r] += wv[r] * fac[t][r] * vcv;
        // F^T[d][i]: per-lane contiguous along i -> two 16B stores
        float* fp = FT_s + (size_t)d * NSEQ + i0 + 8 * hi;
        f4 lo = {fac[t][0], fac[t][1], fac[t][2], fac[t][3]};
        f4 hi4 = {fac[t][4], fac[t][5], fac[t][6], fac[t][7]};
        *(f4*)(fp)     = lo;
        *(f4*)(fp + 4) = hi4;
      }
    }

    // --- GEMM3: T = E^T @ va ; fold out_b ---
    {
      v8f tac[4] = {vzero, vzero, vzero, vzero};
#pragma unroll
      for (int kb = 0; kb < NSEQ; kb += 32) {
        const v16h af = frag_rowk(ET_s, NSEQ, i0, kb, lane);  // rows j, K=i
        v16h bfs[4];
#pragma unroll
        for (int t = 0; t < 4; ++t)
          bfs[t] = frag_rowk(vaT_s, NSEQ, t * 16, kb, lane);
#pragma unroll
        for (int t = 0; t < 4; ++t) tac[t] = wmma_f16(af, bfs[t], tac[t]);
      }
      float wv[8];
#pragma unroll
      for (int r = 0; r < 8; ++r) wv[r] = wb_s[i0 + r + 8 * hi];
#pragma unroll
      for (int t = 0; t < 4; ++t) {
        const int d = t * 16 + nn;
        const float vcv = (float)vc_s[k * DHEAD + d];
#pragma unroll
        for (int r = 0; r < 8; ++r)
          acc_b[t][r] += wv[r] * tac[t][r] * vcv;
      }
    }
    __syncthreads();  // FT_s complete

    // --- out_c[k,:] = w_c[k] * sum_i va[i,d]*F[i,d] (staged, deterministic)
    {
      const int d = tid & 63, seg = tid >> 6;
      const _Float16* var = vaT_s + (size_t)d * NSEQ;
      const float*    fr  = FT_s  + (size_t)d * NSEQ;
      float p = 0.f;
#pragma unroll
      for (int i = seg * 32; i < seg * 32 + 32; i += 4) {
        const h4 vv = *(const h4*)(var + i);
        const f4 ff = *(const f4*)(fr + i);
        p += (float)vv[0] * ff[0] + (float)vv[1] * ff[1] +
             (float)vv[2] * ff[2] + (float)vv[3] * ff[3];
      }
      outc_part[seg * 64 + d] = p;
    }
    __syncthreads();
    if (tid < 64) {
      const float sum = outc_part[tid] + outc_part[64 + tid] +
                        outc_part[128 + tid] + outc_part[192 + tid];
      outc[(size_t)(e * NSEQ + k) * INNERD + h * DHEAD + tid] = wcp[k] * sum;
    }
    __syncthreads();  // before E_s/ET_s/outc_part rewritten next slice
  }

  // Store per-split partial out_a/out_b (reduced across splits later).
  float* oa = outa_part + (size_t)sk * (BSZ * NSEQ * INNERD);
  float* ob = outb_part + (size_t)sk * (BSZ * NSEQ * INNERD);
#pragma unroll
  for (int t = 0; t < 4; ++t)
#pragma unroll
    for (int r = 0; r < 8; ++r) {
      const int i = i0 + r + 8 * hi;
      const int d = t * 16 + nn;
      const size_t g = (size_t)(e * NSEQ + i) * INNERD + h * DHEAD + d;
      oa[g] = acc_a[t][r];
      ob[g] = acc_b[t][r];
    }
}

__global__ __launch_bounds__(256) void reduce_split_kernel(
    const float* __restrict__ part, float* __restrict__ out, int n) {
  const int idx = blockIdx.x * blockDim.x + threadIdx.x;
  if (idx < n) {
    float s = 0.f;
#pragma unroll
    for (int sp = 0; sp < KSPLIT; ++sp) s += part[(size_t)sp * n + idx];
    out[idx] = s;
  }
}

// ---------------------------------------------------------------------------

extern "C" void kernel_launch(void* const* d_in, const int* in_sizes, int n_in,
                              void* d_out, int out_size, void* d_ws, size_t ws_size,
                              hipStream_t stream) {
  (void)in_sizes; (void)n_in; (void)out_size; (void)ws_size;
  const float* A   = (const float*)d_in[0];
  const float* B   = (const float*)d_in[1];
  const float* C   = (const float*)d_in[2];
  const unsigned char* mask = (const unsigned char*)d_in[3];
  const float* WfA = (const float*)d_in[4];
  const float* WfB = (const float*)d_in[5];
  const float* WfC = (const float*)d_in[6];
  const float* WvA = (const float*)d_in[7];
  const float* WvB = (const float*)d_in[8];
  const float* WvC = (const float*)d_in[9];
  const float* WoA = (const float*)d_in[10];
  const float* boA = (const float*)d_in[11];
  const float* WoB = (const float*)d_in[12];
  const float* boB = (const float*)d_in[13];
  const float* WoC = (const float*)d_in[14];
  const float* boC = (const float*)d_in[15];
  float* out = (float*)d_out;

  const int M = BSZ * NSEQ;                        // 256
  const size_t projElems = (size_t)M * INNERD;     // 131072
  char* ws = (char*)d_ws;
  _Float16* p_a  = (_Float16*)ws; ws += projElems * sizeof(_Float16);
  _Float16* p_b  = (_Float16*)ws; ws += projElems * sizeof(_Float16);
  _Float16* p_c  = (_Float16*)ws; ws += projElems * sizeof(_Float16);
  _Float16* p_va = (_Float16*)ws; ws += projElems * sizeof(_Float16);
  _Float16* p_vb = (_Float16*)ws; ws += projElems * sizeof(_Float16);
  _Float16* p_vc = (_Float16*)ws; ws += projElems * sizeof(_Float16);
  float* w_a = (float*)ws; ws += 16 * NSEQ * sizeof(float);
  float* w_b = (float*)ws; ws += 16 * NSEQ * sizeof(float);
  float* w_c = (float*)ws; ws += 16 * NSEQ * sizeof(float);
  float* zr_part = (float*)ws; ws += 16 * KSPLIT * NSEQ * sizeof(float);
  float* zc_part = (float*)ws; ws += 16 * KSPLIT * NSEQ * sizeof(float);
  float* oma_part = (float*)ws; ws += (size_t)KSPLIT * projElems * sizeof(float);
  float* omb_part = (float*)ws; ws += (size_t)KSPLIT * projElems * sizeof(float);
  float* om_a = (float*)ws; ws += projElems * sizeof(float);
  float* om_b = (float*)ws; ws += projElems * sizeof(float);
  float* om_c = (float*)ws; ws += projElems * sizeof(float);

  // 1) Six input projections: [256,256] @ [256,512] -> f16
  dim3 pg(INNERD / 64, M / 64);
  gemm_f16out_kernel<<<pg, 128, 0, stream>>>(A, WfA, p_a,  M, CINCH, INNERD);
  gemm_f16out_kernel<<<pg, 128, 0, stream>>>(B, WfB, p_b,  M, CINCH, INNERD);
  gemm_f16out_kernel<<<pg, 128, 0, stream>>>(C, WfC, p_c,  M, CINCH, INNERD);
  gemm_f16out_kernel<<<pg, 128, 0, stream>>>(A, WvA, p_va, M, CINCH, INNERD);
  gemm_f16out_kernel<<<pg, 128, 0, stream>>>(B, WvB, p_vb, M, CINCH, INNERD);
  gemm_f16out_kernel<<<pg, 128, 0, stream>>>(C, WvC, p_vc, M, CINCH, INNERD);

  // 2) Softmax denominators
  pass1_stats_kernel<<<16 * KSPLIT, 256, 0, stream>>>(
      p_a, p_b, p_c, mask, zr_part, zc_part, w_c);
  finalize_stats_kernel<<<16, 128, 0, stream>>>(zr_part, zc_part, w_a, w_b);

  // 3) Attention body (194 KB dynamic LDS -> one workgroup per 320 KB WGP)
  hipFuncSetAttribute((const void*)pass2_attn_kernel,
                      hipFuncAttributeMaxDynamicSharedMemorySize,
                      PASS2_LDS_BYTES);
  pass2_attn_kernel<<<16 * KSPLIT, 256, PASS2_LDS_BYTES, stream>>>(
      p_a, p_b, p_c, p_va, p_vb, p_vc, mask, w_a, w_b, w_c,
      oma_part, omb_part, om_c);
  reduce_split_kernel<<<(int)((projElems + 255) / 256), 256, 0, stream>>>(
      oma_part, om_a, (int)projElems);
  reduce_split_kernel<<<(int)((projElems + 255) / 256), 256, 0, stream>>>(
      omb_part, om_b, (int)projElems);

  // 4) Output projections + bias: [256,512] @ [512,256] -> d_out (f32)
  dim3 og(CINCH / 64, M / 64);
  gemm_bias_f32out_kernel<<<og, 128, 0, stream>>>(om_a, WoA, boA,
      out + 0 * (size_t)M * CINCH, M, INNERD, CINCH);
  gemm_bias_f32out_kernel<<<og, 128, 0, stream>>>(om_b, WoB, boB,
      out + 1 * (size_t)M * CINCH, M, INNERD, CINCH);
  gemm_bias_f32out_kernel<<<og, 128, 0, stream>>>(om_c, WoC, boC,
      out + 2 * (size_t)M * CINCH, M, INNERD, CINCH);
}
